// ArcFace_39041252721261
// MI455X (gfx1250) — matrix operand order repaired and checked by reading, not compiled, not run on hardware
//
#include <hip/hip_runtime.h>
#include <math.h>
#include <stdint.h>

typedef __attribute__((ext_vector_type(16))) _Float16 v16h;
typedef __attribute__((ext_vector_type(8)))  _Float16 f16x8;
typedef __attribute__((ext_vector_type(8)))  float    v8f;
typedef int v4i __attribute__((__vector_size__(4 * sizeof(int))));

#define PI_F 3.14159265358979f
#define ARC_SCALE 64.0f
#define ARC_MARGIN 0.5f
#define ARC_CLAMP 1e-7f

// ---- CDNA5 async global->LDS DMA availability probes ----
#if defined(__has_builtin)
#  if __has_builtin(__builtin_amdgcn_global_load_async_to_lds_b128)
#    define HAVE_ASYNC_LDS 1
#  endif
#  if __has_builtin(__builtin_amdgcn_s_wait_asynccnt)
#    define HAVE_WAIT_ASYNC 1
#  endif
#endif
#ifndef HAVE_ASYNC_LDS
#  define HAVE_ASYNC_LDS 0
#endif
#ifndef HAVE_WAIT_ASYNC
#  define HAVE_WAIT_ASYNC 0
#endif

#if HAVE_ASYNC_LDS
__device__ __forceinline__ void asyncCopy16B(const _Float16* g, _Float16* l) {
    // Param types per clang diagnostic: (v4i __device__*, v4i __shared__*, Ii, Ii).
    // Generic LDS pointer: low 32 bits are the wave-relative LDS byte offset.
    __builtin_amdgcn_global_load_async_to_lds_b128(
        (__attribute__((address_space(1))) v4i*)(uintptr_t)g,
        (__attribute__((address_space(3))) v4i*)(unsigned)(uintptr_t)l,
        0, 0);
}
__device__ __forceinline__ void waitAsyncLe3() {
#if HAVE_WAIT_ASYNC
    __builtin_amdgcn_s_wait_asynccnt(3);
#else
    asm volatile("s_wait_asynccnt 0x3" ::: "memory");
#endif
}
__device__ __forceinline__ void waitAsync0() {
#if HAVE_WAIT_ASYNC
    __builtin_amdgcn_s_wait_asynccnt(0);
#else
    asm volatile("s_wait_asynccnt 0x0" ::: "memory");
#endif
}
#endif

__device__ __forceinline__ float waveReduceSum(float v) {
    for (int off = 16; off > 0; off >>= 1) v += __shfl_xor(v, off, 32);
    return v;
}

__device__ __forceinline__ v16h combine16(f16x8 lo, f16x8 hi) {
    return __builtin_shufflevector(lo, hi, 0,1,2,3,4,5,6,7,8,9,10,11,12,13,14,15);
}

__device__ __forceinline__ float clipCos(float c) {
    return fminf(fmaxf(c, -1.0f + ARC_CLAMP), 1.0f - ARC_CLAMP);
}

// Per-row: sum of squares -> inv norm; also convert row to f16 (unnormalized).
__global__ __launch_bounds__(128) void prep_rows_kernel(
    const float* __restrict__ src, _Float16* __restrict__ dst,
    float* __restrict__ inv, int ncols) {
    int row = blockIdx.x;
    int t = threadIdx.x;                 // 128 threads = 4 waves
    const float* s = src + (size_t)row * ncols;
    _Float16* d = dst + (size_t)row * ncols;
    float ss = 0.f;
    for (int i = t; i < ncols; i += 128) {
        float x = s[i];
        ss += x * x;
        d[i] = (_Float16)x;
    }
    ss = waveReduceSum(ss);
    __shared__ float red[4];
    if ((t & 31) == 0) red[t >> 5] = ss;
    __syncthreads();
    if (t == 0) {
        float tot = red[0] + red[1] + red[2] + red[3];
        inv[row] = 1.0f / fmaxf(sqrtf(tot), 1e-12f);
    }
}

// Exact f32 GT-column handling: cos_gt, keep mask, margin-adjusted logit.
__global__ __launch_bounds__(128) void gt_kernel(
    const float* __restrict__ emb, const float* __restrict__ w,
    const int* __restrict__ gt, const float* __restrict__ einv,
    const float* __restrict__ winv, float* __restrict__ rowAdj,
    float* __restrict__ rowLgt, float* __restrict__ rowKeep, int ncols) {
    int row = blockIdx.x;
    int t = threadIdx.x;                 // 128 threads
    int g = gt[row];
    const float* e = emb + (size_t)row * ncols;
    const float* wg = w + (size_t)g * ncols;
    float dot = 0.f;
    for (int i = t; i < ncols; i += 128) dot += e[i] * wg[i];
    dot = waveReduceSum(dot);
    __shared__ float red[4];
    if ((t & 31) == 0) red[t >> 5] = dot;
    __syncthreads();
    if (t == 0) {
        float total = red[0] + red[1] + red[2] + red[3];
        float cosg = clipCos(total * einv[row] * winv[g]);
        float th = acosf(cosg);
        rowKeep[row] = (th + ARC_MARGIN <= PI_F) ? 1.f : 0.f;
        rowAdj[row]  = cosf(th + ARC_MARGIN) * ARC_SCALE;
        rowLgt[row]  = cosg * ARC_SCALE;
    }
}

__global__ void zero_kernel(float* __restrict__ p, int n) {
    int i = blockIdx.x * blockDim.x + threadIdx.x;
    if (i < n) p[i] = 0.f;
}

// Fused GEMM + streaming sum(exp(64*cos - 64)) per row.
// Block tile: 64(M) x 128(N), 256 threads = 8 waves arranged 4(M) x 2(N).
// Wave computes 16x64 via 4 accumulators. K stepped by 32 through
// double-buffered LDS with software pipelining (async DMA to LDS if available).
__global__ __launch_bounds__(256) void gemm_softmax_kernel(
    const _Float16* __restrict__ e16, const _Float16* __restrict__ w16,
    const float* __restrict__ einv, const float* __restrict__ winv,
    float* __restrict__ Z, int Nrows, int Ccls, int E) {
    __shared__ _Float16 As[2][64][40];    // pad to 40 hw: 80B row stride (16B aligned)
    __shared__ _Float16 Bs[2][128][40];

    int tid  = threadIdx.x;
    int lane = tid & 31;
    int wave = tid >> 5;
    int wm   = wave & 3;              // M subtile 0..3
    int wn   = wave >> 2;             // N half   0..1
    int half = lane >> 4;
    int l16  = lane & 15;

    int m0 = blockIdx.x * 64;
    int n0 = blockIdx.y * 128;

    int lr = tid >> 2;                // 0..63 staging row
    int lc = (tid & 3) * 8;           // 0,8,16,24 halfwords

    int ar  = m0 + lr;       if (ar  >= Nrows) ar  = Nrows - 1;
    int br0 = n0 + lr;       if (br0 >= Ccls)  br0 = Ccls - 1;
    int br1 = n0 + 64 + lr;  if (br1 >= Ccls)  br1 = Ccls - 1;
    const _Float16* aSrc  = e16 + (size_t)ar  * E + lc;
    const _Float16* bSrc0 = w16 + (size_t)br0 * E + lc;
    const _Float16* bSrc1 = w16 + (size_t)br1 * E + lc;

    v8f c0 = {}, c1 = {}, c2 = {}, c3 = {};

    int am = wm * 16 + l16;           // A fragment row within tile
    int bn = wn * 64 + l16;           // first B fragment row within tile

    const int S = E >> 5;             // K stages of 32

    auto computeStage = [&](int buf) {
        f16x8 alo = *(const f16x8*)&As[buf][am][half * 8];
        f16x8 ahi = *(const f16x8*)&As[buf][am][16 + half * 8];
        v16h a = combine16(alo, ahi);
        #pragma unroll
        for (int sub = 0; sub < 4; ++sub) {
            f16x8 blo = *(const f16x8*)&Bs[buf][bn + sub * 16][half * 8];
            f16x8 bhi = *(const f16x8*)&Bs[buf][bn + sub * 16][16 + half * 8];
            v16h b = combine16(blo, bhi);
            v8f* cp = (sub == 0) ? &c0 : (sub == 1) ? &c1 : (sub == 2) ? &c2 : &c3;
            *cp = __builtin_amdgcn_wmma_f32_16x16x32_f16(false, a, false, b,
                                                         (short)0, *cp, false, false);
        }
    };

#if HAVE_ASYNC_LDS
    auto issueStage = [&](int s, int buf) {
        int kk = s << 5;
        asyncCopy16B(aSrc  + kk, &As[buf][lr][lc]);
        asyncCopy16B(bSrc0 + kk, &Bs[buf][lr][lc]);
        asyncCopy16B(bSrc1 + kk, &Bs[buf][lr + 64][lc]);
    };
    issueStage(0, 0);
    for (int s = 0; s < S; ++s) {
        int buf = s & 1;
        bool more = (s + 1 < S);
        if (more) issueStage(s + 1, buf ^ 1);   // overlap DMA with compute
        if (more) waitAsyncLe3(); else waitAsync0();
        __syncthreads();                         // this buffer fully landed
        computeStage(buf);
        __syncthreads();                         // safe to DMA-overwrite next time
    }
#else
    // register-staged double-buffered pipeline
    f16x8 ra  = *(const f16x8*)(aSrc);
    f16x8 rb0 = *(const f16x8*)(bSrc0);
    f16x8 rb1 = *(const f16x8*)(bSrc1);
    for (int s = 0; s < S; ++s) {
        int buf = s & 1;
        *(f16x8*)&As[buf][lr][lc]      = ra;
        *(f16x8*)&Bs[buf][lr][lc]      = rb0;
        *(f16x8*)&Bs[buf][lr + 64][lc] = rb1;
        __syncthreads();
        if (s + 1 < S) {                        // prefetch next stage over compute
            int kk = (s + 1) << 5;
            ra  = *(const f16x8*)(aSrc  + kk);
            rb0 = *(const f16x8*)(bSrc0 + kk);
            rb1 = *(const f16x8*)(bSrc1 + kk);
        }
        computeStage(buf);
        __syncthreads();
    }
#endif

    // Epilogue: cos = acc * einv[m] * winv[n]; accumulate exp(64*cos - 64) per row.
    int   nG[4];
    float wi[4];
    #pragma unroll
    for (int sub = 0; sub < 4; ++sub) {
        nG[sub] = n0 + wn * 64 + sub * 16 + l16;
        wi[sub] = (nG[sub] < Ccls) ? winv[nG[sub]] : 0.f;
    }
    #pragma unroll
    for (int v = 0; v < 8; ++v) {
        int m = m0 + wm * 16 + half * 8 + v;   // C-layout: VGPR v -> M = v + 8*half
        float s = 0.f;
        if (m < Nrows) {
            float ei = einv[m];
            if (nG[0] < Ccls) s += expf(ARC_SCALE * clipCos(c0[v] * ei * wi[0]) - ARC_SCALE);
            if (nG[1] < Ccls) s += expf(ARC_SCALE * clipCos(c1[v] * ei * wi[1]) - ARC_SCALE);
            if (nG[2] < Ccls) s += expf(ARC_SCALE * clipCos(c2[v] * ei * wi[2]) - ARC_SCALE);
            if (nG[3] < Ccls) s += expf(ARC_SCALE * clipCos(c3[v] * ei * wi[3]) - ARC_SCALE);
        }
        for (int off = 8; off > 0; off >>= 1) s += __shfl_xor(s, off, 16);
        if (l16 == 0 && m < Nrows) atomicAdd(&Z[m], s);
    }
}

// nll = 64 + log(Z - exp(lgt-64) + exp(adj-64)) - adj; loss = sum(nll*keep)/max(cnt,1)
__global__ __launch_bounds__(256) void finalize_kernel(
    const float* __restrict__ Z, const float* __restrict__ rowAdj,
    const float* __restrict__ rowLgt, const float* __restrict__ rowKeep,
    int Nrows, float* __restrict__ out) {
    int t = threadIdx.x;   // 256 threads, single block
    float accL = 0.f, accC = 0.f;
    for (int r = t; r < Nrows; r += 256) {
        float adj = rowAdj[r];
        float Zs = Z[r] - expf(rowLgt[r] - ARC_SCALE) + expf(adj - ARC_SCALE);
        Zs = fmaxf(Zs, 1e-37f);
        float nll = ARC_SCALE + logf(Zs) - adj;
        float k = rowKeep[r];
        accL += nll * k;
        accC += k;
    }
    accL = waveReduceSum(accL);
    accC = waveReduceSum(accC);
    __shared__ float rl[8], rc[8];
    if ((t & 31) == 0) { rl[t >> 5] = accL; rc[t >> 5] = accC; }
    __syncthreads();
    if (t == 0) {
        float L = 0.f, Cc = 0.f;
        for (int i = 0; i < 8; ++i) { L += rl[i]; Cc += rc[i]; }
        out[0] = L / fmaxf(Cc, 1.f);
    }
}

extern "C" void kernel_launch(void* const* d_in, const int* in_sizes, int n_in,
                              void* d_out, int out_size, void* d_ws, size_t ws_size,
                              hipStream_t stream) {
    const float* emb = (const float*)d_in[0];
    const int*   gt  = (const int*)d_in[1];
    const float* wgt = (const float*)d_in[2];

    const int Nrows = in_sizes[1];                 // 1024
    const int E     = in_sizes[0] / Nrows;         // 512
    const int Ccls  = in_sizes[2] / E;             // 100000

    // workspace layout (256B-aligned sections)
    char* ws = (char*)d_ws;
    size_t off = 0;
    auto carve = [&](size_t bytes) -> char* {
        char* p = ws + off;
        off = (off + bytes + 255) & ~(size_t)255;
        return p;
    };
    _Float16* w16  = (_Float16*)carve((size_t)Ccls * E * sizeof(_Float16));
    _Float16* e16  = (_Float16*)carve((size_t)Nrows * E * sizeof(_Float16));
    float* winv    = (float*)carve((size_t)Ccls * sizeof(float));
    float* einv    = (float*)carve((size_t)Nrows * sizeof(float));
    float* Zacc    = (float*)carve((size_t)Nrows * sizeof(float));
    float* rowAdj  = (float*)carve((size_t)Nrows * sizeof(float));
    float* rowLgt  = (float*)carve((size_t)Nrows * sizeof(float));
    float* rowKeep = (float*)carve((size_t)Nrows * sizeof(float));
    if (off > ws_size) return;   // workspace too small: refuse to scribble

    float* out = (float*)d_out;

    // 1) convert + inverse norms
    prep_rows_kernel<<<Nrows, 128, 0, stream>>>(emb, e16, einv, E);
    prep_rows_kernel<<<Ccls, 128, 0, stream>>>(wgt, w16, winv, E);

    // 2) exact GT column in f32 (needs einv/winv)
    gt_kernel<<<Nrows, 128, 0, stream>>>(emb, wgt, gt, einv, winv,
                                         rowAdj, rowLgt, rowKeep, E);

    // 3) zero per-row partition accumulators
    zero_kernel<<<(Nrows + 255) / 256, 256, 0, stream>>>(Zacc, Nrows);

    // 4) fused WMMA GEMM + streaming sum-exp (M tiles fastest for B-tile L2 reuse)
    dim3 grid((Nrows + 63) / 64, (Ccls + 127) / 128);
    gemm_softmax_kernel<<<grid, 256, 0, stream>>>(e16, w16, einv, winv,
                                                  Zacc, Nrows, Ccls, E);

    // 5) per-row NLL + masked mean
    finalize_kernel<<<1, 256, 0, stream>>>(Zacc, rowAdj, rowLgt, rowKeep,
                                           Nrows, out);
}
